// TSPDGraphTransformerNetwork_5712306504233
// MI455X (gfx1250) — compile-verified, matrix operand order checked by reading, not run on hardware
//
#include <hip/hip_runtime.h>
#include <hip/hip_bf16.h>

typedef __attribute__((ext_vector_type(16))) _Float16 v16h;
typedef __attribute__((ext_vector_type(8)))  _Float16 v8h;
typedef __attribute__((ext_vector_type(8)))  float    v8f;

#define HIDC 128
#define SCALE_ATT 0.17677669529663687f  // 1/sqrt(32)

#define SHUF16(lo, hi) __builtin_shufflevector(lo, hi, 0, 1, 2, 3, 4, 5, 6, 7, \
                                               8, 9, 10, 11, 12, 13, 14, 15)

// ---------------------------------------------------------------------------
// float atomic max via CAS
// ---------------------------------------------------------------------------
__device__ __forceinline__ void atomicMaxF(float* addr, float val) {
    unsigned int* u = (unsigned int*)addr;
    unsigned int old = *u;
    while (__uint_as_float(old) < val) {
        unsigned int assumed = old;
        old = atomicCAS(u, assumed, __float_as_uint(val));
        if (old == assumed) break;
    }
}

// ---------------------------------------------------------------------------
// Per-layer scratch init: agg=0, amax=-inf, den=0, BN sums=0
// ---------------------------------------------------------------------------
__global__ __launch_bounds__(256)
void layer_init_kernel(float* __restrict__ agg, float* __restrict__ amax,
                       float* __restrict__ den, float* __restrict__ bnsum,
                       float* __restrict__ bnsq, int Nn) {
    int t = blockIdx.x * 256 + threadIdx.x;
    if (t < Nn * HIDC) agg[t] = 0.0f;
    if (t < Nn * 4) { amax[t] = -3.402823466e38f; den[t] = 0.0f; }
    if (t < HIDC) { bnsum[t] = 0.0f; bnsq[t] = 0.0f; }
}

// ---------------------------------------------------------------------------
// Weight prep (per layer): f32 W[K][N] (4 mats) -> f16 Wt[cat_col][K] (transposed)
// plus concatenated f32 bias bcat[512]
// ---------------------------------------------------------------------------
__global__ __launch_bounds__(256)
void prep_weights_kernel(const float* __restrict__ Wq, const float* __restrict__ bq,
                         const float* __restrict__ Wk, const float* __restrict__ bk,
                         const float* __restrict__ Wv, const float* __restrict__ bv,
                         const float* __restrict__ Ws, const float* __restrict__ bs,
                         _Float16* __restrict__ Wt, float* __restrict__ bcat) {
    int t = blockIdx.x * 256 + threadIdx.x;      // over 512*128
    if (t >= 4 * HIDC * HIDC) return;
    int cg = t / HIDC;                           // concat column 0..511
    int k  = t % HIDC;
    int m = cg >> 7, n = cg & 127;
    const float* W = (m == 0) ? Wq : (m == 1) ? Wk : (m == 2) ? Wv : Ws;
    Wt[(size_t)cg * HIDC + k] = (_Float16)W[(size_t)k * HIDC + n];
    if (k == 0) {
        const float* B = (m == 0) ? bq : (m == 1) ? bk : (m == 2) ? bv : bs;
        bcat[cg] = B[n];
    }
}

// ---------------------------------------------------------------------------
// x f32 -> f16 (layer 0 only; later layers get xh from bn_elu)
// ---------------------------------------------------------------------------
__global__ __launch_bounds__(256)
void prep_x_kernel(const float* __restrict__ xin, _Float16* __restrict__ xh, int n) {
    int t = blockIdx.x * 256 + threadIdx.x;
    if (t < n) xh[t] = (_Float16)xin[t];
}

// ---------------------------------------------------------------------------
// Fused q/k/v/skip GEMM. 8 waves/block; wave w computes cols [16w,16w+16) of
// all four 128x128 weight matrices for one 16-row tile. All WMMA operands are
// loaded as two contiguous b128 chunks straight from global f16 buffers
// (fragment layout == memory layout). 16 static v_wmma per wave.
// ---------------------------------------------------------------------------
__global__ __launch_bounds__(256)
void gemm_qkvs_kernel(const _Float16* __restrict__ xh,
                      const _Float16* __restrict__ Wt,
                      const float* __restrict__ bcat,
                      float* __restrict__ qo, float* __restrict__ ko,
                      float* __restrict__ vo, float* __restrict__ so,
                      int nrows) {
    const int tid  = threadIdx.x;
    const int wave = tid >> 5;
    const int lane = tid & 31;
    const int mrow = lane & 15;   // A row / B,C column within tile
    const int half = lane >> 4;
    const int r0   = blockIdx.x * 16;

    // A fragments for the whole K=128, shared across the 4 weight matrices.
    int arow = min(r0 + mrow, nrows - 1);        // clamp keeps EXEC uniform
    const _Float16* Arow = xh + (size_t)arow * HIDC;
    v16h afrag[4];
#pragma unroll
    for (int ks = 0; ks < 4; ++ks) {
        int k0 = ks * 32 + half * 8;
        v8h lo = *(const v8h*)(Arow + k0);
        v8h hi = *(const v8h*)(Arow + k0 + 16);
        afrag[ks] = SHUF16(lo, hi);
    }

    const bool full = (r0 + 16 <= nrows);
#pragma unroll
    for (int m = 0; m < 4; ++m) {
        const int ncol = wave * 16 + mrow;       // column within matrix m
        const int cg   = m * HIDC + ncol;        // concat column
        const _Float16* Brow = Wt + (size_t)cg * HIDC;
        v8f acc = {};
#pragma unroll
        for (int ks = 0; ks < 4; ++ks) {
            int k0 = ks * 32 + half * 8;
            v8h lo = *(const v8h*)(Brow + k0);
            v8h hi = *(const v8h*)(Brow + k0 + 16);
            v16h b = SHUF16(lo, hi);
            acc = __builtin_amdgcn_wmma_f32_16x16x32_f16(
                false, afrag[ks], false, b, (short)0, acc, false, false);
        }
        float bias = bcat[cg];
        float* O = (m == 0) ? qo : (m == 1) ? ko : (m == 2) ? vo : so;
        if (full) {
#pragma unroll
            for (int r = 0; r < 8; ++r)          // C/D: M = r + 8*half
                O[(size_t)(r0 + r + half * 8) * HIDC + ncol] = acc[r] + bias;
        } else {
#pragma unroll
            for (int r = 0; r < 8; ++r) {
                int grow = r0 + r + half * 8;
                if (grow < nrows) O[(size_t)grow * HIDC + ncol] = acc[r] + bias;
            }
        }
    }
}

// ---------------------------------------------------------------------------
// Edge pass 1: alpha[e,h] = <q[dst], k[src]+e_emb> * scale ; segment max
// ---------------------------------------------------------------------------
__global__ __launch_bounds__(256)
void edge_alpha_kernel(const long long* __restrict__ ei,
                       const float* __restrict__ q, const float* __restrict__ kb,
                       const float* __restrict__ ea,
                       const float* __restrict__ We, const float* __restrict__ be,
                       float* __restrict__ alpha, float* __restrict__ amax, int Ee) {
    int t = blockIdx.x * 256 + threadIdx.x;
    if (t >= Ee * 4) return;
    int e = t >> 2, h = t & 3;
    int src = (int)ei[e];
    int dst = (int)ei[Ee + e];
    float a0 = ea[(size_t)e * 2], a1 = ea[(size_t)e * 2 + 1];
    int c0 = h * 32;
    const float4* q4 = (const float4*)(q  + (size_t)dst * HIDC + c0);
    const float4* k4 = (const float4*)(kb + (size_t)src * HIDC + c0);
    const float4* w0 = (const float4*)(We + c0);
    const float4* w1 = (const float4*)(We + HIDC + c0);
    const float4* b4 = (const float4*)(be + c0);
    float acc = 0.0f;
#pragma unroll
    for (int i = 0; i < 8; ++i) {
        float4 qq = q4[i], kk = k4[i], wa = w0[i], wb = w1[i], bb = b4[i];
        acc += qq.x * (kk.x + a0 * wa.x + a1 * wb.x + bb.x);
        acc += qq.y * (kk.y + a0 * wa.y + a1 * wb.y + bb.y);
        acc += qq.z * (kk.z + a0 * wa.z + a1 * wb.z + bb.z);
        acc += qq.w * (kk.w + a0 * wa.w + a1 * wb.w + bb.w);
    }
    acc *= SCALE_ATT;
    alpha[t] = acc;
    atomicMaxF(&amax[(size_t)dst * 4 + h], acc);
}

// ---------------------------------------------------------------------------
// Edge pass 2: ex = exp(alpha - amax[dst]); segment sum
// ---------------------------------------------------------------------------
__global__ __launch_bounds__(256)
void edge_expsum_kernel(const long long* __restrict__ ei,
                        float* __restrict__ alpha, const float* __restrict__ amax,
                        float* __restrict__ den, int Ee) {
    int t = blockIdx.x * 256 + threadIdx.x;
    if (t >= Ee * 4) return;
    int e = t >> 2, h = t & 3;
    int dst = (int)ei[Ee + e];
    float ex = __expf(alpha[t] - amax[(size_t)dst * 4 + h]);
    alpha[t] = ex;
    atomicAdd(&den[(size_t)dst * 4 + h], ex);
}

// ---------------------------------------------------------------------------
// Edge pass 3: scatter-add (v[src]+e_emb) * softmax weight into agg[dst]
// ---------------------------------------------------------------------------
__global__ __launch_bounds__(256)
void edge_message_kernel(const long long* __restrict__ ei,
                         const float* __restrict__ vb, const float* __restrict__ ea,
                         const float* __restrict__ We, const float* __restrict__ be,
                         const float* __restrict__ alpha, const float* __restrict__ den,
                         float* __restrict__ agg, int Ee) {
    int t = blockIdx.x * 256 + threadIdx.x;
    if (t >= Ee * 4) return;
    int e = t >> 2, h = t & 3;
    int src = (int)ei[e];
    int dst = (int)ei[Ee + e];
    float w = alpha[t] / (den[(size_t)dst * 4 + h] + 1e-16f);
    float a0 = ea[(size_t)e * 2], a1 = ea[(size_t)e * 2 + 1];
    int c0 = h * 32;
    const float4* v4 = (const float4*)(vb + (size_t)src * HIDC + c0);
    const float4* w0 = (const float4*)(We + c0);
    const float4* w1 = (const float4*)(We + HIDC + c0);
    const float4* b4 = (const float4*)(be + c0);
    float* ob = agg + (size_t)dst * HIDC + c0;
#pragma unroll
    for (int i = 0; i < 8; ++i) {
        float4 vv = v4[i], wa = w0[i], wb = w1[i], bb = b4[i];
        atomicAdd(ob + i * 4 + 0, (vv.x + a0 * wa.x + a1 * wb.x + bb.x) * w);
        atomicAdd(ob + i * 4 + 1, (vv.y + a0 * wa.y + a1 * wb.y + bb.y) * w);
        atomicAdd(ob + i * 4 + 2, (vv.z + a0 * wa.z + a1 * wb.z + bb.z) * w);
        atomicAdd(ob + i * 4 + 3, (vv.w + a0 * wa.w + a1 * wb.w + bb.w) * w);
    }
}

// ---------------------------------------------------------------------------
// Beta-gated skip + BN statistics (block-local LDS reduction, then global)
// ---------------------------------------------------------------------------
__global__ __launch_bounds__(256)
void node_gate_kernel(float* __restrict__ agg, const float* __restrict__ xr,
                      const float* __restrict__ Wb,
                      float* __restrict__ bnsum, float* __restrict__ bnsq, int Nn) {
    __shared__ float ssum[HIDC];
    __shared__ float ssq[HIDC];
    int tid = threadIdx.x;
    if (tid < HIDC) { ssum[tid] = 0.0f; ssq[tid] = 0.0f; }
    __syncthreads();
    int n = blockIdx.x * 256 + tid;
    if (n < Nn) {
        size_t base = (size_t)n * HIDC;
        const float4* o4 = (const float4*)(agg + base);
        const float4* r4 = (const float4*)(xr + base);
        const float4* w0 = (const float4*)(Wb);
        const float4* w1 = (const float4*)(Wb + HIDC);
        const float4* w2 = (const float4*)(Wb + 2 * HIDC);
        float sg = 0.0f;
        for (int i = 0; i < 32; ++i) {
            float4 o = o4[i], r = r4[i], a = w0[i], b = w1[i], c = w2[i];
            sg += o.x * a.x + r.x * b.x + (o.x - r.x) * c.x;
            sg += o.y * a.y + r.y * b.y + (o.y - r.y) * c.y;
            sg += o.z * a.z + r.z * b.z + (o.z - r.z) * c.z;
            sg += o.w * a.w + r.w * b.w + (o.w - r.w) * c.w;
        }
        float g = 1.0f / (1.0f + __expf(-sg));
        float4* oo4 = (float4*)(agg + base);
        for (int i = 0; i < 32; ++i) {
            float4 o = o4[i], r = r4[i], bl;
            bl.x = g * r.x + (1.0f - g) * o.x;
            bl.y = g * r.y + (1.0f - g) * o.y;
            bl.z = g * r.z + (1.0f - g) * o.z;
            bl.w = g * r.w + (1.0f - g) * o.w;
            oo4[i] = bl;
            int c = i * 4;
            atomicAdd(&ssum[c + 0], bl.x); atomicAdd(&ssq[c + 0], bl.x * bl.x);
            atomicAdd(&ssum[c + 1], bl.y); atomicAdd(&ssq[c + 1], bl.y * bl.y);
            atomicAdd(&ssum[c + 2], bl.z); atomicAdd(&ssq[c + 2], bl.z * bl.z);
            atomicAdd(&ssum[c + 3], bl.w); atomicAdd(&ssq[c + 3], bl.w * bl.w);
        }
    }
    __syncthreads();
    if (tid < HIDC) {
        atomicAdd(&bnsum[tid], ssum[tid]);
        atomicAdd(&bnsq[tid], ssq[tid]);
    }
}

// ---------------------------------------------------------------------------
// BatchNorm (training stats, biased var) + ELU -> new features (f32 + f16)
// ---------------------------------------------------------------------------
__global__ __launch_bounds__(256)
void bn_elu_kernel(const float* __restrict__ agg, float* __restrict__ xout,
                   _Float16* __restrict__ xh,
                   const float* __restrict__ bnsum, const float* __restrict__ bnsq,
                   const float* __restrict__ gamma, const float* __restrict__ beta,
                   int Nn) {
    int t = blockIdx.x * 256 + threadIdx.x;
    if (t >= Nn * HIDC) return;
    int c = t & 127;
    float invN = 1.0f / (float)Nn;
    float mu  = bnsum[c] * invN;
    float var = bnsq[c] * invN - mu * mu;
    float y = (agg[t] - mu) * rsqrtf(var + 1e-5f) * gamma[c] + beta[c];
    float r = (y > 0.0f) ? y : (__expf(y) - 1.0f);
    xout[t] = r;
    xh[t] = (_Float16)r;
}

// ---------------------------------------------------------------------------
// Readout: per-graph mean pool fused with final linear
// ---------------------------------------------------------------------------
__global__ __launch_bounds__(256)
void pool_init_kernel(float* __restrict__ psum, float* __restrict__ pcnt, int Gg) {
    int t = blockIdx.x * 256 + threadIdx.x;
    if (t < Gg) { psum[t] = 0.0f; pcnt[t] = 0.0f; }
}

__global__ __launch_bounds__(256)
void pool_accum_kernel(const float* __restrict__ xf, const long long* __restrict__ batch,
                       const float* __restrict__ Wout,
                       float* __restrict__ psum, float* __restrict__ pcnt, int Nn) {
    int n = blockIdx.x * 256 + threadIdx.x;
    if (n >= Nn) return;
    const float4* x4 = (const float4*)(xf + (size_t)n * HIDC);
    const float4* w4 = (const float4*)(Wout);
    float s = 0.0f;
#pragma unroll
    for (int i = 0; i < 32; ++i) {
        float4 x = x4[i], w = w4[i];
        s += x.x * w.x + x.y * w.y + x.z * w.z + x.w * w.w;
    }
    int g = (int)batch[n];
    atomicAdd(&psum[g], s);
    atomicAdd(&pcnt[g], 1.0f);
}

__global__ __launch_bounds__(256)
void pool_final_kernel(const float* __restrict__ psum, const float* __restrict__ pcnt,
                       const float* __restrict__ bout, const float* __restrict__ obias,
                       float* __restrict__ out, int Gg) {
    int g = blockIdx.x * 256 + threadIdx.x;
    if (g >= Gg) return;
    out[g] = psum[g] / fmaxf(pcnt[g], 1.0f) + bout[0] + obias[0];
}

// ---------------------------------------------------------------------------
extern "C" void kernel_launch(void* const* d_in, const int* in_sizes, int n_in,
                              void* d_out, int out_size, void* d_ws, size_t ws_size,
                              hipStream_t stream) {
    const float*     x_in  = (const float*)d_in[0];
    const long long* ei    = (const long long*)d_in[1];
    const float*     ea    = (const float*)d_in[2];
    const long long* batch = (const long long*)d_in[3];
    const float* Wq   = (const float*)d_in[4];
    const float* bq   = (const float*)d_in[5];
    const float* Wk   = (const float*)d_in[6];
    const float* bk   = (const float*)d_in[7];
    const float* Wv   = (const float*)d_in[8];
    const float* bv   = (const float*)d_in[9];
    const float* We   = (const float*)d_in[10];
    const float* be   = (const float*)d_in[11];
    const float* Wsk  = (const float*)d_in[12];
    const float* bsk  = (const float*)d_in[13];
    const float* Wbt  = (const float*)d_in[14];
    const float* gam  = (const float*)d_in[15];
    const float* bet  = (const float*)d_in[16];
    const float* Wout = (const float*)d_in[17];
    const float* bout = (const float*)d_in[18];
    const float* obia = (const float*)d_in[19];

    const int Nn = in_sizes[0] / HIDC;   // 50000
    const int Ee = in_sizes[2] / 2;      // 800000
    const int Gg = out_size;             // 50
    const int L  = in_sizes[5] / HIDC;   // 3 layers

    // workspace layout (float units)
    float* ws = (float*)d_ws;
    size_t o = 0;
    float* qb    = ws + o; o += (size_t)Nn * HIDC;
    float* kb    = ws + o; o += (size_t)Nn * HIDC;
    float* vb    = ws + o; o += (size_t)Nn * HIDC;
    float* sb    = ws + o; o += (size_t)Nn * HIDC;
    float* agg   = ws + o; o += (size_t)Nn * HIDC;
    float* xbuf  = ws + o; o += (size_t)Nn * HIDC;
    float* alpha = ws + o; o += (size_t)Ee * 4;
    float* amax  = ws + o; o += (size_t)Nn * 4;
    float* den   = ws + o; o += (size_t)Nn * 4;
    float* bnsum = ws + o; o += HIDC;
    float* bnsq  = ws + o; o += HIDC;
    float* psum  = ws + o; o += (size_t)Gg;
    float* pcnt  = ws + o; o += (size_t)Gg + 2;    // +2 keeps next region 16B aligned
    _Float16* xh = (_Float16*)(ws + o); o += (size_t)Nn * HIDC / 2;
    _Float16* Wt = (_Float16*)(ws + o); o += (size_t)4 * HIDC * HIDC / 2;
    float* bcat  = ws + o; o += 4 * HIDC;
    (void)n_in; (void)ws_size;

    const int nTileBlocks = (Nn + 15) / 16;
    const int nNodeChan   = (Nn * HIDC + 255) / 256;
    const int nEdgeHead   = (Ee * 4 + 255) / 256;
    const int nNode       = (Nn + 255) / 256;

    prep_x_kernel<<<nNodeChan, 256, 0, stream>>>(x_in, xh, Nn * HIDC);

    for (int l = 0; l < L; ++l) {
        const float* Wq_l  = Wq  + (size_t)l * HIDC * HIDC;
        const float* Wk_l  = Wk  + (size_t)l * HIDC * HIDC;
        const float* Wv_l  = Wv  + (size_t)l * HIDC * HIDC;
        const float* Ws_l  = Wsk + (size_t)l * HIDC * HIDC;
        const float* bq_l  = bq  + (size_t)l * HIDC;
        const float* bk_l  = bk  + (size_t)l * HIDC;
        const float* bv_l  = bv  + (size_t)l * HIDC;
        const float* bs_l  = bsk + (size_t)l * HIDC;
        const float* We_l  = We  + (size_t)l * 2 * HIDC;
        const float* be_l  = be  + (size_t)l * HIDC;
        const float* Wb_l  = Wbt + (size_t)l * 3 * HIDC;
        const float* gam_l = gam + (size_t)l * HIDC;
        const float* bet_l = bet + (size_t)l * HIDC;

        layer_init_kernel<<<nNodeChan, 256, 0, stream>>>(agg, amax, den, bnsum, bnsq, Nn);

        prep_weights_kernel<<<(4 * HIDC * HIDC + 255) / 256, 256, 0, stream>>>(
            Wq_l, bq_l, Wk_l, bk_l, Wv_l, bv_l, Ws_l, bs_l, Wt, bcat);

        gemm_qkvs_kernel<<<nTileBlocks, 256, 0, stream>>>(
            xh, Wt, bcat, qb, kb, vb, sb, Nn);

        edge_alpha_kernel<<<nEdgeHead, 256, 0, stream>>>(
            ei, qb, kb, ea, We_l, be_l, alpha, amax, Ee);

        edge_expsum_kernel<<<nEdgeHead, 256, 0, stream>>>(ei, alpha, amax, den, Ee);

        edge_message_kernel<<<nEdgeHead, 256, 0, stream>>>(
            ei, vb, ea, We_l, be_l, alpha, den, agg, Ee);

        node_gate_kernel<<<nNode, 256, 0, stream>>>(agg, sb, Wb_l, bnsum, bnsq, Nn);

        bn_elu_kernel<<<nNodeChan, 256, 0, stream>>>(agg, xbuf, xh, bnsum, bnsq,
                                                     gam_l, bet_l, Nn);
    }

    pool_init_kernel<<<(Gg + 255) / 256, 256, 0, stream>>>(psum, pcnt, Gg);
    pool_accum_kernel<<<nNode, 256, 0, stream>>>(xbuf, batch, Wout, psum, pcnt, Nn);
    pool_final_kernel<<<(Gg + 255) / 256, 256, 0, stream>>>(psum, pcnt, bout, obia,
                                                            (float*)d_out, Gg);
}
